// MinkGlobalEnc_78958678769886
// MI455X (gfx1250) — matrix operand order repaired and checked by reading, not compiled, not run on hardware
//
#include <hip/hip_runtime.h>

typedef __attribute__((ext_vector_type(16))) _Float16 v16h;
typedef __attribute__((ext_vector_type(8)))  _Float16 v8h;
typedef __attribute__((ext_vector_type(8)))  float    v8f;

// ---------------------------------------------------------------------------
// Implicit-GEMM 3D convolution on WMMA (f16 operands, f32 accumulate),
// 2x2 register-blocked: each wave computes a 32(M) x 32(N) tile with 4
// accumulators, so every A/B fragment fetched from L2 is used twice.
//   X : NDHWC f16 activations [B, IE^3, Cin]   (Cin multiple of 32)
//   Wh: f16 weights [tap][Cout][Cin]
//   zp: zero-filled f16 page (>= Cin entries) for out-of-bounds taps
//   Y : NDHWC f32 output [B, OE^3, Cout]
// Inner 32-K chunk: 8x global_load_b128 + 4x v_wmma  (32 B/lane per WMMA).
// ---------------------------------------------------------------------------
__global__ __launch_bounds__(32)
void mink_conv3d_wmma(const _Float16* __restrict__ X, const _Float16* __restrict__ Wh,
                      const _Float16* __restrict__ zp, float* __restrict__ Y,
                      int B, int Cin, int Cout,
                      int IE, int OE, int ks, int stride, int pad)
{
    const int lane = threadIdx.x & 31;
    const int grp  = lane >> 4;
    const int l15  = lane & 15;
    const int ovox = OE * OE * OE;
    const int Mtot = B * ovox;
    const int m0   = blockIdx.x * 32;
    const int n0   = blockIdx.y * 32;

    // ---- two M sub-tiles: decode this lane's output voxel for each ----
    int  bb[2], id0[2], ih0[2], iw0[2];
    bool vmr[2];
#pragma unroll
    for (int mi = 0; mi < 2; ++mi) {
        int mrow = m0 + mi * 16 + l15;
        vmr[mi] = (mrow < Mtot);
        int mm = vmr[mi] ? mrow : 0;
        int b  = mm / ovox;
        int r  = mm - b * ovox;
        int od = r / (OE * OE);
        int r2 = r - od * (OE * OE);
        int oh = r2 / OE;
        int ow = r2 - oh * OE;
        bb[mi]  = b;
        id0[mi] = od * stride - pad;
        ih0[mi] = oh * stride - pad;
        iw0[mi] = ow * stride - pad;
    }

    // ---- two N sub-tiles ----
    int  ncol[2], ncl[2];
    bool vn[2];
#pragma unroll
    for (int ni = 0; ni < 2; ++ni) {
        ncol[ni] = n0 + ni * 16 + l15;
        vn[ni]   = (ncol[ni] < Cout);
        ncl[ni]  = vn[ni] ? ncol[ni] : 0;
    }

    v8f acc[2][2] = {};
    for (int kd = 0; kd < ks; ++kd)
    for (int kh = 0; kh < ks; ++kh)
    for (int kw = 0; kw < ks; ++kw) {
        const int tap = (kd * ks + kh) * ks + kw;

        const _Float16* src[2];
#pragma unroll
        for (int mi = 0; mi < 2; ++mi) {
            const int  id = id0[mi] + kd, ih = ih0[mi] + kh, iw = iw0[mi] + kw;
            const bool inb = vmr[mi] && (unsigned)id < (unsigned)IE &&
                                        (unsigned)ih < (unsigned)IE &&
                                        (unsigned)iw < (unsigned)IE;
            const _Float16* base = inb ?
                X + ((((size_t)bb[mi] * IE + id) * IE + ih) * IE + iw) * (size_t)Cin
                : zp;                                  // OOB taps read zeros
            src[mi] = base + grp * 16;
        }
        const _Float16* wr0 = Wh + ((size_t)tap * Cout + ncl[0]) * Cin + grp * 16;
        const _Float16* wr1 = Wh + ((size_t)tap * Cout + ncl[1]) * Cin + grp * 16;

        for (int c0 = 0; c0 < Cin; c0 += 32) {
            const v8h a0a = *(const v8h*)(src[0] + c0);
            const v8h a0b = *(const v8h*)(src[0] + c0 + 8);
            const v8h a1a = *(const v8h*)(src[1] + c0);
            const v8h a1b = *(const v8h*)(src[1] + c0 + 8);
            const v8h b0a = *(const v8h*)(wr0 + c0);
            const v8h b0b = *(const v8h*)(wr0 + c0 + 8);
            const v8h b1a = *(const v8h*)(wr1 + c0);
            const v8h b1b = *(const v8h*)(wr1 + c0 + 8);
            v16h af0, af1, bf0, bf1;
#pragma unroll
            for (int h = 0; h < 8; ++h) {
                af0[h] = a0a[h]; af0[h + 8] = a0b[h];
                af1[h] = a1a[h]; af1[h + 8] = a1b[h];
                bf0[h] = b0a[h]; bf0[h + 8] = b0b[h];
                bf1[h] = b1a[h]; bf1[h + 8] = b1b[h];
            }
            acc[0][0] = __builtin_amdgcn_wmma_f32_16x16x32_f16(
                            false, af0, false, bf0, (short)0, acc[0][0], false, false);
            acc[0][1] = __builtin_amdgcn_wmma_f32_16x16x32_f16(
                            false, af0, false, bf1, (short)0, acc[0][1], false, false);
            acc[1][0] = __builtin_amdgcn_wmma_f32_16x16x32_f16(
                            false, af1, false, bf0, (short)0, acc[1][0], false, false);
            acc[1][1] = __builtin_amdgcn_wmma_f32_16x16x32_f16(
                            false, af1, false, bf1, (short)0, acc[1][1], false, false);
        }
    }

#pragma unroll
    for (int mi = 0; mi < 2; ++mi) {
#pragma unroll
        for (int rr = 0; rr < 8; ++rr) {
            int M = m0 + mi * 16 + rr + grp * 8;
            if (M < Mtot) {
                if (vn[0]) Y[(size_t)M * Cout + ncol[0]] = acc[mi][0][rr];
                if (vn[1]) Y[(size_t)M * Cout + ncol[1]] = acc[mi][1][rr];
            }
        }
    }
}

// OIDHW f32 -> [tap][Cout][CinPad] f16 (zero-fill ci >= Cin).
__global__ void wxform(const float* __restrict__ Wsrc, _Float16* __restrict__ Wdst,
                       int Cout, int Cin, int CinPad, int KV)
{
    size_t i = (size_t)blockIdx.x * blockDim.x + threadIdx.x;
    size_t tot = (size_t)KV * Cout * CinPad;
    if (i >= tot) return;
    int    ci = (int)(i % (size_t)CinPad);
    size_t t  = i / (size_t)CinPad;
    int    o  = (int)(t % (size_t)Cout);
    int    tp = (int)(t / (size_t)Cout);
    float  v  = (ci < Cin) ? Wsrc[((size_t)o * Cin + ci) * KV + tp] : 0.0f;
    Wdst[i] = (_Float16)v;
}

// NCDHW [B,3,vox] f32 -> NDHWC [B,vox,32] f16, channels 3..31 zero-filled.
__global__ void feats_pad(const float* __restrict__ src, _Float16* __restrict__ dst,
                          int vox, int B)
{
    size_t i = (size_t)blockIdx.x * blockDim.x + threadIdx.x;
    size_t tot = ((size_t)B * vox) << 5;
    if (i >= tot) return;
    int    c = (int)(i & 31);
    size_t v = i >> 5;                       // b*vox + p
    size_t b = v / (size_t)vox, p = v - b * vox;
    dst[i] = (c < 3) ? (_Float16)src[(b * 3 + c) * (size_t)vox + p] : (_Float16)0.0f;
}

// Zero-fill an f16 region (zero page for OOB conv taps).
__global__ void zfill_h(_Float16* __restrict__ p, int n)
{
    int i = blockIdx.x * blockDim.x + threadIdx.x;
    if (i < n) p[i] = (_Float16)0.0f;
}

// Per-channel masked sums over NDHWC: s1[c]=sum(x*m), s2[c]=sum(x^2*m).
__global__ void bn_stats(const float* __restrict__ Y, const float* __restrict__ Msk,
                         float* __restrict__ s1, float* __restrict__ s2,
                         int C, size_t nvox)
{
    const int c = blockIdx.x;
    float a = 0.0f, q = 0.0f;
    for (size_t v = threadIdx.x; v < nvox; v += blockDim.x) {
        float val = Y[v * C + c] * Msk[v];
        a += val;
        q += val * val;       // m in {0,1}
    }
    __shared__ float sa[256], sb[256];
    sa[threadIdx.x] = a; sb[threadIdx.x] = q;
    __syncthreads();
    for (int s = blockDim.x >> 1; s > 0; s >>= 1) {
        if ((int)threadIdx.x < s) {
            sa[threadIdx.x] += sa[threadIdx.x + s];
            sb[threadIdx.x] += sb[threadIdx.x + s];
        }
        __syncthreads();
    }
    if (threadIdx.x == 0) { s1[c] = sa[0]; s2[c] = sb[0]; }
}

// Occupied-voxel count for one mask level, clamped to >= 1.
__global__ void mask_count(const float* __restrict__ m, float* __restrict__ cnt, int n)
{
    float a = 0.0f;
    for (int i = threadIdx.x; i < n; i += blockDim.x) a += m[i];
    __shared__ float s[256];
    s[threadIdx.x] = a;
    __syncthreads();
    for (int st = blockDim.x >> 1; st > 0; st >>= 1) {
        if ((int)threadIdx.x < st) s[threadIdx.x] += s[threadIdx.x + st];
        __syncthreads();
    }
    if (threadIdx.x == 0) cnt[0] = fmaxf(s[0], 1.0f);
}

// y = (x-mean)*m*rsqrt(var+eps) [+ add] [relu].
// NDHWC in; writes f32 NDHWC (plus optional f16 twin) or f32 NCDHW. C = 1<<lc.
__global__ void bn_apply(const float* Yin, const float* Msk,
                         const float* s1, const float* s2, const float* cnt,
                         const float* addsrc, float* out, _Float16* outh,
                         int C, int lc, int vox, size_t nvox, int do_relu, int out_nchw)
{
    size_t i = (size_t)blockIdx.x * blockDim.x + threadIdx.x;
    size_t total = nvox << lc;
    if (i >= total) return;
    int    c = (int)(i & (size_t)(C - 1));
    size_t v = i >> lc;                       // global voxel = b*vox + p
    float m    = Msk[v];
    float n    = cnt[0];
    float mean = s1[c] / n;
    float var  = fmaxf(s2[c] / n - mean * mean, 0.0f);
    float val  = (Yin[i] - mean) * m * rsqrtf(var + 1e-5f);
    if (addsrc) val += addsrc[i];
    if (do_relu) val = fmaxf(val, 0.0f);
    if (out_nchw) {
        size_t b = v / (size_t)vox, p = v - b * (size_t)vox;
        out[((size_t)b * C + c) * vox + p] = val;
    } else {
        out[i] = val;
        if (outh) outh[i] = (_Float16)val;
    }
}

// 2x2x2 max-pool of the occupancy mask ([B,vox] flat). (D,H,W) = OUTPUT dims.
__global__ void downmask_k(const float* __restrict__ m, float* __restrict__ mo,
                           int B, int D, int H, int W)
{
    size_t i = (size_t)blockIdx.x * blockDim.x + threadIdx.x;
    size_t tot = (size_t)B * D * H * W;
    if (i >= tot) return;
    int w = (int)(i % W);
    int h = (int)((i / W) % H);
    int d = (int)((i / ((size_t)W * H)) % D);
    int b = (int)(i / ((size_t)W * H * D));
    int IH = 2 * H, IW = 2 * W, IDm = 2 * D;
    const float* mb = m + (size_t)b * IDm * IH * IW;
    float v = 0.0f;
    for (int dz = 0; dz < 2; ++dz)
        for (int dy = 0; dy < 2; ++dy)
            for (int dx = 0; dx < 2; ++dx)
                v = fmaxf(v, mb[((size_t)(2 * d + dz) * IH + (2 * h + dy)) * IW +
                                (2 * w + dx)]);
    mo[i] = v;
}

extern "C" void kernel_launch(void* const* d_in, const int* in_sizes, int n_in,
                              void* d_out, int out_size, void* d_ws, size_t ws_size,
                              hipStream_t stream)
{
    (void)in_sizes; (void)n_in; (void)out_size; (void)ws_size;

    const float* feats = (const float*)d_in[0];
    const float* mask0 = (const float*)d_in[1];

    // f32 workspace region
    float* bufA = (float*)d_ws;               // (2,64^3,32) NDHWC raw/normalized
    float* bufB = bufA + (size_t)16777216;
    float* bufC = bufB + (size_t)16777216;    // (2,32^3,32) residual temporaries
    float* m1   = bufC + (size_t)2097152;
    float* m2   = m1 + 65536;
    float* m3   = m2 + 8192;
    float* m4   = m3 + 1024;
    float* s1   = m4 + 128;
    float* s2   = s1 + 256;
    float* cnt  = s2 + 256;                   // 8 floats (keeps 32B alignment)
    // f16 workspace region
    _Float16* zp = (_Float16*)(cnt + 8);      // 512-entry zero page
    _Float16* hA = zp + 512;                  // f16 twin of bufA
    _Float16* hB = hA + (size_t)16777216;     // f16 twin of bufB
    _Float16* wbase = hB + (size_t)16777216;  // f16 weight arena (~17 MB)

    const int B = 2;
    size_t wcur = 0;

    auto prep = [&](int idx, int Cout, int Cin, int CinPad, int ks) -> const _Float16* {
        const float* Wsrc = (const float*)d_in[idx];
        int KV = ks * ks * ks;
        size_t tot = (size_t)KV * Cout * CinPad;
        _Float16* dst = wbase + wcur;
        wcur += (tot + 15) & ~(size_t)15;     // keep 32B alignment
        wxform<<<dim3((unsigned)((tot + 255) / 256)), dim3(256), 0, stream>>>(
            Wsrc, dst, Cout, Cin, CinPad, KV);
        return dst;
    };

    zfill_h<<<dim3(2), dim3(256), 0, stream>>>(zp, 512);

    const _Float16* pws0 = prep(2, 32, 3, 32, 3);
    const _Float16* pws1 = prep(3, 32, 32, 32, 3);
    const _Float16* pd1  = prep(4, 32, 32, 32, 2);
    const _Float16* pa11 = prep(5, 32, 32, 32, 3);
    const _Float16* pb11 = prep(6, 32, 32, 32, 3);
    const _Float16* pa12 = prep(7, 32, 32, 32, 3);
    const _Float16* pb12 = prep(8, 32, 32, 32, 3);
    const _Float16* pd2  = prep(9, 32, 32, 32, 2);
    const _Float16* pa21 = prep(10, 64, 32, 32, 3);
    const _Float16* pb21 = prep(11, 64, 64, 64, 3);
    const _Float16* pk2  = prep(12, 64, 32, 32, 1);
    const _Float16* pa22 = prep(13, 64, 64, 64, 3);
    const _Float16* pb22 = prep(14, 64, 64, 64, 3);
    const _Float16* pd3  = prep(15, 64, 64, 64, 2);
    const _Float16* pa31 = prep(16, 128, 64, 64, 3);
    const _Float16* pb31 = prep(17, 128, 128, 128, 3);
    const _Float16* pk3  = prep(18, 128, 64, 64, 1);
    const _Float16* pa32 = prep(19, 128, 128, 128, 3);
    const _Float16* pb32 = prep(20, 128, 128, 128, 3);
    const _Float16* pd4  = prep(21, 128, 128, 128, 2);
    const _Float16* pa41 = prep(22, 256, 128, 128, 3);
    const _Float16* pb41 = prep(23, 256, 256, 256, 3);
    const _Float16* pk4  = prep(24, 256, 128, 128, 1);
    const _Float16* pa42 = prep(25, 256, 256, 256, 3);
    const _Float16* pb42 = prep(26, 256, 256, 256, 3);

    auto conv = [&](const _Float16* X, const _Float16* Wh, float* Y,
                    int Cin, int Cout, int IE, int OE, int ks, int stride, int pad) {
        int Mtot = B * OE * OE * OE;
        dim3 g((unsigned)((Mtot + 31) / 32), (unsigned)((Cout + 31) / 32), 1);
        mink_conv3d_wmma<<<g, dim3(32), 0, stream>>>(
            X, Wh, zp, Y, B, Cin, Cout, IE, OE, ks, stride, pad);
    };
    auto stats = [&](const float* Y, const float* M, int C, int OE) {
        size_t nv = (size_t)B * OE * OE * OE;
        bn_stats<<<dim3((unsigned)C), dim3(256), 0, stream>>>(Y, M, s1, s2, C, nv);
    };
    auto apply = [&](const float* Y, const float* M, const float* add, float* outp,
                     _Float16* outh, int C, int lc, int OE, int lvl, int relu, int nchw) {
        size_t nv = (size_t)B * OE * OE * OE;
        size_t tot = nv << lc;
        bn_apply<<<dim3((unsigned)((tot + 255) / 256)), dim3(256), 0, stream>>>(
            Y, M, s1, s2, cnt + lvl, add, outp, outh, C, lc, OE * OE * OE, nv, relu, nchw);
    };
    auto down = [&](const float* Min, float* Mout, int OE) {
        size_t tot = (size_t)B * OE * OE * OE;
        downmask_k<<<dim3((unsigned)((tot + 255) / 256)), dim3(256), 0, stream>>>(
            Min, Mout, B, OE, OE, OE);
    };
    auto count = [&](const float* M, int OE, int lvl) {
        mask_count<<<dim3(1), dim3(256), 0, stream>>>(M, cnt + lvl, B * OE * OE * OE);
    };

    // ---- stem @ 64^3 : pad feats NCDHW(3ch,f32) -> NDHWC(32ch,f16) in hB ----
    count(mask0, 64, 0);
    {
        size_t tot = ((size_t)B * 262144) << 5;
        feats_pad<<<dim3((unsigned)((tot + 255) / 256)), dim3(256), 0, stream>>>(
            feats, hB, 262144, B);
    }
    conv(hB, pws0, bufA, 32, 32, 64, 64, 3, 1, 1);
    stats(bufA, mask0, 32, 64);  apply(bufA, mask0, nullptr, bufA, hA, 32, 5, 64, 0, 1, 0);
    conv(hA, pws1, bufB, 32, 32, 64, 64, 3, 1, 1);
    stats(bufB, mask0, 32, 64);  apply(bufB, mask0, nullptr, bufB, hB, 32, 5, 64, 0, 1, 0);

    // ---- stage 1: 64^3 -> 32^3, 32ch, identity skips; x: hB -> bufA/hA ----
    down(mask0, m1, 32);  count(m1, 32, 1);
    conv(hB, pd1, bufA, 32, 32, 64, 32, 2, 2, 0);
    stats(bufA, m1, 32, 32);  apply(bufA, m1, nullptr, bufA, hA, 32, 5, 32, 1, 1, 0);
    conv(hA, pa11, bufB, 32, 32, 32, 32, 3, 1, 1);
    stats(bufB, m1, 32, 32);  apply(bufB, m1, nullptr, bufB, hB, 32, 5, 32, 1, 1, 0);
    conv(hB, pb11, bufC, 32, 32, 32, 32, 3, 1, 1);
    stats(bufC, m1, 32, 32);  apply(bufC, m1, bufA, bufA, hA, 32, 5, 32, 1, 1, 0);
    conv(hA, pa12, bufB, 32, 32, 32, 32, 3, 1, 1);
    stats(bufB, m1, 32, 32);  apply(bufB, m1, nullptr, bufB, hB, 32, 5, 32, 1, 1, 0);
    conv(hB, pb12, bufC, 32, 32, 32, 32, 3, 1, 1);
    stats(bufC, m1, 32, 32);  apply(bufC, m1, bufA, bufA, hA, 32, 5, 32, 1, 1, 0);

    // ---- stage 2: 32^3 -> 16^3, 32 -> 64ch (skip k2); x ends in bufB/hB ----
    down(m1, m2, 16);  count(m2, 16, 2);
    conv(hA, pd2, bufB, 32, 32, 32, 16, 2, 2, 0);
    stats(bufB, m2, 32, 16);  apply(bufB, m2, nullptr, bufB, hB, 32, 5, 16, 2, 1, 0);
    conv(hB, pa21, bufA, 32, 64, 16, 16, 3, 1, 1);
    stats(bufA, m2, 64, 16);  apply(bufA, m2, nullptr, bufA, hA, 64, 6, 16, 2, 1, 0);
    conv(hA, pb21, bufC, 64, 64, 16, 16, 3, 1, 1);
    stats(bufC, m2, 64, 16);  apply(bufC, m2, nullptr, bufC, nullptr, 64, 6, 16, 2, 0, 0);
    conv(hB, pk2, bufA, 32, 64, 16, 16, 1, 1, 0);
    stats(bufA, m2, 64, 16);  apply(bufA, m2, bufC, bufB, hB, 64, 6, 16, 2, 1, 0);
    conv(hB, pa22, bufA, 64, 64, 16, 16, 3, 1, 1);
    stats(bufA, m2, 64, 16);  apply(bufA, m2, nullptr, bufA, hA, 64, 6, 16, 2, 1, 0);
    conv(hA, pb22, bufC, 64, 64, 16, 16, 3, 1, 1);
    stats(bufC, m2, 64, 16);  apply(bufC, m2, bufB, bufB, hB, 64, 6, 16, 2, 1, 0);

    // ---- stage 3: 16^3 -> 8^3, 64 -> 128ch (skip k3); x ends in bufA/hA ----
    down(m2, m3, 8);  count(m3, 8, 3);
    conv(hB, pd3, bufA, 64, 64, 16, 8, 2, 2, 0);
    stats(bufA, m3, 64, 8);  apply(bufA, m3, nullptr, bufA, hA, 64, 6, 8, 3, 1, 0);
    conv(hA, pa31, bufB, 64, 128, 8, 8, 3, 1, 1);
    stats(bufB, m3, 128, 8);  apply(bufB, m3, nullptr, bufB, hB, 128, 7, 8, 3, 1, 0);
    conv(hB, pb31, bufC, 128, 128, 8, 8, 3, 1, 1);
    stats(bufC, m3, 128, 8);  apply(bufC, m3, nullptr, bufC, nullptr, 128, 7, 8, 3, 0, 0);
    conv(hA, pk3, bufB, 64, 128, 8, 8, 1, 1, 0);
    stats(bufB, m3, 128, 8);  apply(bufB, m3, bufC, bufA, hA, 128, 7, 8, 3, 1, 0);
    conv(hA, pa32, bufB, 128, 128, 8, 8, 3, 1, 1);
    stats(bufB, m3, 128, 8);  apply(bufB, m3, nullptr, bufB, hB, 128, 7, 8, 3, 1, 0);
    conv(hB, pb32, bufC, 128, 128, 8, 8, 3, 1, 1);
    stats(bufC, m3, 128, 8);  apply(bufC, m3, bufA, bufA, hA, 128, 7, 8, 3, 1, 0);

    // ---- stage 4: 8^3 -> 4^3, 128 -> 256ch (skip k4); out -> d_out NCDHW ----
    down(m3, m4, 4);  count(m4, 4, 4);
    conv(hA, pd4, bufB, 128, 128, 8, 4, 2, 2, 0);
    stats(bufB, m4, 128, 4);  apply(bufB, m4, nullptr, bufB, hB, 128, 7, 4, 4, 1, 0);
    conv(hB, pa41, bufA, 128, 256, 4, 4, 3, 1, 1);
    stats(bufA, m4, 256, 4);  apply(bufA, m4, nullptr, bufA, hA, 256, 8, 4, 4, 1, 0);
    conv(hA, pb41, bufC, 256, 256, 4, 4, 3, 1, 1);
    stats(bufC, m4, 256, 4);  apply(bufC, m4, nullptr, bufC, nullptr, 256, 8, 4, 4, 0, 0);
    conv(hB, pk4, bufA, 128, 256, 4, 4, 1, 1, 0);
    stats(bufA, m4, 256, 4);  apply(bufA, m4, bufC, bufB, hB, 256, 8, 4, 4, 1, 0);
    conv(hB, pa42, bufA, 256, 256, 4, 4, 3, 1, 1);
    stats(bufA, m4, 256, 4);  apply(bufA, m4, nullptr, bufA, hA, 256, 8, 4, 4, 1, 0);
    conv(hA, pb42, bufC, 256, 256, 4, 4, 3, 1, 1);
    stats(bufC, m4, 256, 4);
    apply(bufC, m4, bufB, (float*)d_out, nullptr, 256, 8, 4, 4, 1, 1);
}
